// DSQGAttentionQW_1047972020955
// MI455X (gfx1250) — compile-verified, matrix operand order checked by reading, not run on hardware
//
#include <hip/hip_runtime.h>
#include <hip/hip_bf16.h>

typedef __attribute__((ext_vector_type(16))) __bf16 v16bf;
typedef __attribute__((ext_vector_type(8)))  __bf16 v8bf;
typedef __attribute__((ext_vector_type(8)))  float  v8f;

#define BB   2
#define NN   4096
#define DD   1024
#define HH   16
#define HDIM 64
#define NOFF 44
#define NQS  (HH * NOFF)   /* 704 real qs columns */
#define NQSP 768           /* padded to a multiple of 128 */
#define NEGINF (-1e30f)

// sorted(set(range(33)) | {48,64,96,128,192,256,384,512,768,1024,1536})
__constant__ int c_off[NOFF] = {
    0, 1, 2, 3, 4, 5, 6, 7, 8, 9, 10, 11, 12, 13, 14, 15,
    16, 17, 18, 19, 20, 21, 22, 23, 24, 25, 26, 27, 28, 29, 30, 31,
    32, 48, 64, 96, 128, 192, 256, 384, 512, 768, 1024, 1536};

// ---------------------------------------------------------------------------
// fp32 -> bf16
// ---------------------------------------------------------------------------
__global__ __launch_bounds__(256) void cvt_f32_bf16(const float* __restrict__ in,
                                                    __hip_bfloat16* __restrict__ out,
                                                    long n) {
    long i = (long)blockIdx.x * blockDim.x + threadIdx.x;
    if (i < n) out[i] = __float2bfloat16(in[i]);
}

// ---------------------------------------------------------------------------
// Fold scale_embed into the q-projection (rows >= 704 are zero padding):
//   W2[(h*44+o), din] = sum_hd qkv_w[h*64+hd, din] * se[o, hd]   (bf16)
//   bias2[h*44+o]     = sum_hd qkv_b[h*64+hd]      * se[o, hd]
// ---------------------------------------------------------------------------
__global__ __launch_bounds__(256) void build_w2(const float* __restrict__ qkv_w,
                                                const float* __restrict__ se,
                                                __hip_bfloat16* __restrict__ w2) {
    const int idx = blockIdx.x * 256 + threadIdx.x;
    if (idx >= NQSP * DD) return;
    const int din  = idx & (DD - 1);
    const int rowo = idx >> 10;
    float s = 0.0f;
    if (rowo < NQS) {
        const int h = rowo / NOFF, oi = rowo % NOFF;
        for (int hd = 0; hd < HDIM; ++hd)
            s += qkv_w[(size_t)(h * HDIM + hd) * DD + din] * se[oi * HDIM + hd];
    }
    w2[idx] = __float2bfloat16(s);
}

__global__ __launch_bounds__(256) void build_bias2(const float* __restrict__ qkv_b,
                                                   const float* __restrict__ se,
                                                   float* __restrict__ bias2) {
    const int rowo = blockIdx.x * 256 + threadIdx.x;
    if (rowo >= NQSP) return;
    float s = 0.0f;
    if (rowo < NQS) {
        const int h = rowo / NOFF, oi = rowo % NOFF;
        for (int hd = 0; hd < HDIM; ++hd)
            s += qkv_b[h * HDIM + hd] * se[oi * HDIM + hd];
    }
    bias2[rowo] = s;
}

// ---------------------------------------------------------------------------
// TDM tile staging. Tile = rows x 64 bf16 (128B payload/row).
// D# pad: pad_interval=32 dwords (128B), pad_amount=4 dwords (16B)
//   -> LDS row stride 144B = 36 dwords; 36*k mod 64 hits 16 distinct banks
//      for k=0..15, so 16-lane fragment reads are bank-conflict free.
// ---------------------------------------------------------------------------
#define LDS_ROW_STRIDE 144
#define LDS_TILE_BYTES (128 * LDS_ROW_STRIDE)          /* one 128-row buffer */
#define LDS_TOTAL_BYTES (4 * LDS_TILE_BYTES)           /* A[2] + B[2] = 72KB */

#if defined(__HIP_DEVICE_COMPILE__) && defined(__has_builtin)
#if __has_builtin(__builtin_amdgcn_tensor_load_to_lds)
#define USE_TDM 1
#endif
#endif
#ifndef USE_TDM
#define USE_TDM 0
#endif

#if USE_TDM
typedef unsigned int u32x4 __attribute__((ext_vector_type(4)));
typedef int          i32x8 __attribute__((ext_vector_type(8)));
typedef int          i32x4 __attribute__((ext_vector_type(4)));

__device__ __forceinline__ void tdm_load_tile(const void* gsrc, unsigned int lds_off,
                                              int rows, int K) {
    const unsigned long long ga = (unsigned long long)(uintptr_t)gsrc;
    u32x4 g0;
    g0[0] = 1u;                                           // count=1 (valid user D#)
    g0[1] = lds_off;                                      // lds_addr (bytes)
    g0[2] = (unsigned int)ga;                             // global_addr[95:64]
    g0[3] = (unsigned int)((ga >> 32) & 0x01FFFFFFu)      // global_addr[120:96]
            | (2u << 30);                                 // type=2 ("image")
    i32x8 g1;
    g1[0] = (1 << 16)                                     // data_size = 2 bytes
          | (1 << 20)                                     // pad_enable
          | (4 << 22)                                     // pad_interval: 32 dwords
          | (3 << 25);                                    // pad_amount: 4 dwords
    g1[1] = (K & 0xFFFF) << 16;                           // tensor_dim0 lo16 (bits 79:48)
    g1[2] = ((K >> 16) & 0xFFFF);                         // tensor_dim0 hi | tensor_dim1 lo (dim1=1<<20)
    g1[3] = 16 | (64 << 16);                              // tensor_dim1 hi16 (=16) | tile_dim0=64
    g1[4] = rows & 0xFFFF;                                // tile_dim1 | tile_dim2=0
    g1[5] = K;                                            // tensor_dim0_stride lo32 (elem units)
    g1[6] = 0;                                            // stride hi16 | dim1_stride lo16
    g1[7] = 0;
#if __clang_major__ >= 23
    __builtin_amdgcn_tensor_load_to_lds(g0, g1, (i32x4)0, (i32x4)0, (i32x8)0, 0);
#else
    __builtin_amdgcn_tensor_load_to_lds(g0, g1, (i32x4)0, (i32x4)0, 0);
#endif
}
#endif

// fallback / host-pass path: cooperative copy into the same padded layout
__device__ __forceinline__ void copy_tile(const __hip_bfloat16* gsrc, unsigned char* dst,
                                          int rows, int K, int tid) {
    const int nchunks = rows * 8;  // 8 x 16B per 128B row
    for (int c = tid; c < nchunks; c += 256) {
        const int r = c >> 3, q = c & 7;
        const float4 d = *(const float4*)((const char*)gsrc + (size_t)r * K * 2 + q * 16);
        *(float4*)(dst + r * LDS_ROW_STRIDE + q * 16) = d;
    }
}

__device__ __forceinline__ v16bf load_frag_lds(const unsigned char* p) {
    union {
        v16bf v;
        v8bf  h[2];
    } u;
    u.h[0] = *(const v8bf*)(const void*)(p);
    u.h[1] = *(const v8bf*)(const void*)(p + 32);
    return u.v;
}

__device__ __forceinline__ v8f wmma_bf16(v16bf a, v16bf b, v8f c) {
    return __builtin_amdgcn_wmma_f32_16x16x32_bf16(false, a, false, b, (short)0, c,
                                                   false, false);
}

// ---------------------------------------------------------------------------
// WMMA bf16 GEMM with TDM double-buffered LDS staging.
//   C[M,Nc] = A[M,K] * W[Nc,K]^T + bias[Nc]
//   MODE 0: Cf = val (fp32);  MODE 1: Cbf = bf16(mulsrc * sigmoid(val))
// Block tile 128(M) x 128(N), K-step 64; 8 waves as 2(M) x 4(N); each wave
// computes a 64x32 tile with 8 accumulators (each A fragment reused twice, each B fragment 4x).
// LDS map (dynamic): A buffers at {0,1}*TILE, B buffers at {2,3}*TILE.
// All call sites: M%128==0, Nc%128==0, K%64==0.
// ---------------------------------------------------------------------------
template <int MODE>
__global__ __launch_bounds__(256) void gemm_bf16_wmma(
    const __hip_bfloat16* __restrict__ A, const __hip_bfloat16* __restrict__ W,
    const float* __restrict__ bias, float* __restrict__ Cf,
    const float* __restrict__ mulsrc, __hip_bfloat16* __restrict__ Cbf,
    int M, int Nc, int K) {
    extern __shared__ unsigned char ldsbuf[];

    const int tid   = threadIdx.x;
    const int lane  = tid & 31;
    const int wave  = tid >> 5;
    const int waveM = wave >> 2;          // 0..1 (64-row halves)
    const int waveN = wave & 3;           // 0..3 (32-col quarters)
    const int l15   = lane & 15;
    const int kh    = (lane >> 4) << 3;   // 0 or 8 (K sub-chunk select)

    const __hip_bfloat16* Ablk = A + (size_t)(blockIdx.y * 128) * K;
    const __hip_bfloat16* Wblk = W + (size_t)(blockIdx.x * 128) * K;

    v8f acc[4][2];
#pragma unroll
    for (int t = 0; t < 4; ++t)
#pragma unroll
        for (int j = 0; j < 2; ++j) acc[t][j] = (v8f){};

    const int nsteps = K >> 6;

    // prologue: stage step 0 into buffer 0
#if USE_TDM
    if (wave == 0) {
        tdm_load_tile(Ablk, (unsigned int)(uintptr_t)(ldsbuf), 128, K);
        tdm_load_tile(Wblk, (unsigned int)(uintptr_t)(ldsbuf + 2 * LDS_TILE_BYTES), 128, K);
    }
#else
    copy_tile(Ablk, ldsbuf, 128, K, tid);
    copy_tile(Wblk, ldsbuf + 2 * LDS_TILE_BYTES, 128, K, tid);
#endif

    for (int i = 0; i < nsteps; ++i) {
        const int buf = i & 1, nxt = buf ^ 1;
        unsigned char* curA = ldsbuf + buf * LDS_TILE_BYTES;
        unsigned char* curB = ldsbuf + (2 + buf) * LDS_TILE_BYTES;
        unsigned char* nxtA = ldsbuf + nxt * LDS_TILE_BYTES;
        unsigned char* nxtB = ldsbuf + (2 + nxt) * LDS_TILE_BYTES;
#if USE_TDM
        if (wave == 0) {
            if (i + 1 < nsteps) {
                tdm_load_tile(Ablk + (i + 1) * 64, (unsigned int)(uintptr_t)nxtA, 128, K);
                tdm_load_tile(Wblk + (i + 1) * 64, (unsigned int)(uintptr_t)nxtB, 128, K);
                __builtin_amdgcn_s_wait_tensorcnt(2);  // oldest 2 (current buf) done
            } else {
                __builtin_amdgcn_s_wait_tensorcnt(0);
            }
        }
#else
        if (i + 1 < nsteps) {
            copy_tile(Ablk + (i + 1) * 64, nxtA, 128, K, tid);
            copy_tile(Wblk + (i + 1) * 64, nxtB, 128, K, tid);
        }
#endif
        __syncthreads();  // current buffer visible to all waves
        const unsigned char* bA = curA + (waveM * 64 + l15) * LDS_ROW_STRIDE;
        const unsigned char* bB = curB + (waveN * 32 + l15) * LDS_ROW_STRIDE;
#pragma unroll
        for (int ks = 0; ks < 2; ++ks) {
            const int cb = ks * 64 + kh * 2;  // byte offset of this 16x32 K-window
            // issue ALL fragment loads of this sub-step before any WMMA so the
            // scheduler can overlap ds latency with matrix issue
            v16bf b0 = load_frag_lds(bB + cb);
            v16bf b1 = load_frag_lds(bB + 16 * LDS_ROW_STRIDE + cb);
            v16bf a0 = load_frag_lds(bA + cb);
            v16bf a1 = load_frag_lds(bA + 16 * LDS_ROW_STRIDE + cb);
            v16bf a2 = load_frag_lds(bA + 32 * LDS_ROW_STRIDE + cb);
            v16bf a3 = load_frag_lds(bA + 48 * LDS_ROW_STRIDE + cb);
            acc[0][0] = wmma_bf16(a0, b0, acc[0][0]);
            acc[0][1] = wmma_bf16(a0, b1, acc[0][1]);
            acc[1][0] = wmma_bf16(a1, b0, acc[1][0]);
            acc[1][1] = wmma_bf16(a1, b1, acc[1][1]);
            acc[2][0] = wmma_bf16(a2, b0, acc[2][0]);
            acc[2][1] = wmma_bf16(a2, b1, acc[2][1]);
            acc[3][0] = wmma_bf16(a3, b0, acc[3][0]);
            acc[3][1] = wmma_bf16(a3, b1, acc[3][1]);
        }
        __syncthreads();  // done reading buf before it is restaged
    }

    // C/D layout: lanes 0-15 -> N=lane, M=r; lanes 16-31 -> M=r+8
    const int mBase  = blockIdx.y * 128 + waveM * 64;
    const int ncBase = blockIdx.x * 128 + waveN * 32 + l15;
    const int rOff   = (lane < 16) ? 0 : 8;
#pragma unroll
    for (int j = 0; j < 2; ++j) {
        const int   nc = ncBase + j * 16;
        const float bv = bias[nc];
#pragma unroll
        for (int t = 0; t < 4; ++t) {
#pragma unroll
            for (int r = 0; r < 8; ++r) {
                const int    m   = mBase + t * 16 + rOff + r;
                const float  val = acc[t][j][r] + bv;
                const size_t idx = (size_t)m * Nc + nc;
                if (MODE == 0) {
                    Cf[idx] = val;
                } else {
                    const float g = 1.0f / (1.0f + __expf(-val));
                    Cbf[idx] = __float2bfloat16(mulsrc[idx] * g);
                }
            }
        }
    }
}

// ---------------------------------------------------------------------------
// Dyadic-offset attention. One wave32 per (b,h,n), split in two 16-lane groups:
// group g = lane>>4 handles offsets oi = g, g+2, ...; each lane owns 4 of the
// 64 head dims -> dot reduce = 4 shuffles per offset pair. qs is precomputed
// (WMMA GEMM, stride NQSP). Two-way online-softmax merge via shfl_xor(16).
// ---------------------------------------------------------------------------
__global__ __launch_bounds__(256) void attn_dyadic(
    const float* __restrict__ qkv, const float* __restrict__ qs,
    const float* __restrict__ pos_bias, const float* __restrict__ if_gain,
    float* __restrict__ outf) {
    const int lane = threadIdx.x & 31;
    const int wave = threadIdx.x >> 5;
    const int row  = blockIdx.x * 8 + wave;  // 0 .. B*H*N-1
    const int n = row & (NN - 1);
    const int h = (row >> 12) & (HH - 1);    // N = 2^12
    const int b = row >> 16;                 // N*H = 2^16
    const int sub = lane & 15;
    const int g   = lane >> 4;

    const size_t tok   = (size_t)(b * NN + n);
    const float4 qq    = *(const float4*)(qkv + tok * 3 * DD + h * HDIM + 4 * sub);
    const float* kbase = qkv + tok * 3 * DD + DD + h * HDIM + 4 * sub;
    const float* qsrow = qs + tok * NQSP + h * NOFF;

    float  m_run = NEGINF, ssum = 0.0f;
    float4 acc = {0.0f, 0.0f, 0.0f, 0.0f};

    for (int t = 0; t < NOFF / 2; ++t) {
        const int oi = 2 * t + g;
        const int o  = c_off[oi];
        float  lg;
        float4 vv = {0.0f, 0.0f, 0.0f, 0.0f};
        if (n >= o) {
            const float* kp = kbase - (size_t)o * (3 * DD);
            const float4 kk = *(const float4*)kp;
            vv = *(const float4*)(kp + DD);
            float pd = qq.x * kk.x + qq.y * kk.y + qq.z * kk.z + qq.w * kk.w;
            pd += __shfl_xor(pd, 1, 32);
            pd += __shfl_xor(pd, 2, 32);
            pd += __shfl_xor(pd, 4, 32);
            pd += __shfl_xor(pd, 8, 32);
            lg = pd * 0.125f * (1.0f + qsrow[oi]) + pos_bias[oi * HH + h];
        } else {
            lg = NEGINF;  // masked; wiped by rescale once a valid offset appears
        }
        const float m_new = fmaxf(m_run, lg);
        const float sc    = __expf(m_run - m_new);
        const float p     = __expf(lg - m_new);
        ssum  = ssum * sc + p;
        acc.x = acc.x * sc + p * vv.x;
        acc.y = acc.y * sc + p * vv.y;
        acc.z = acc.z * sc + p * vv.z;
        acc.w = acc.w * sc + p * vv.w;
        m_run = m_new;
    }

    // merge the two groups (group0 always has offset 0 -> finite max overall)
    const float m_o = __shfl_xor(m_run, 16, 32);
    const float s_o = __shfl_xor(ssum, 16, 32);
    float4      a_o;
    a_o.x = __shfl_xor(acc.x, 16, 32);
    a_o.y = __shfl_xor(acc.y, 16, 32);
    a_o.z = __shfl_xor(acc.z, 16, 32);
    a_o.w = __shfl_xor(acc.w, 16, 32);
    const float m_f = fmaxf(m_run, m_o);
    const float c0  = __expf(m_run - m_f);  // 0 for an all-masked group
    const float c1  = __expf(m_o - m_f);
    const float s_f = ssum * c0 + s_o * c1;
    const float gn  = if_gain[h] / s_f;
    if (g == 0) {
        float4 r;
        r.x = (acc.x * c0 + a_o.x * c1) * gn;
        r.y = (acc.y * c0 + a_o.y * c1) * gn;
        r.z = (acc.z * c0 + a_o.z * c1) * gn;
        r.w = (acc.w * c0 + a_o.w * c1) * gn;
        *(float4*)(outf + tok * DD + h * HDIM + 4 * sub) = r;
    }
}

// ---------------------------------------------------------------------------
extern "C" void kernel_launch(void* const* d_in, const int* in_sizes, int n_in,
                              void* d_out, int out_size, void* d_ws, size_t ws_size,
                              hipStream_t stream) {
    (void)in_sizes; (void)n_in; (void)out_size; (void)ws_size;
    const float* x           = (const float*)d_in[0];
    const float* qkv_w       = (const float*)d_in[1];
    const float* qkv_b       = (const float*)d_in[2];
    const float* out_w       = (const float*)d_in[3];
    const float* out_b       = (const float*)d_in[4];
    const float* gate_w      = (const float*)d_in[5];
    const float* gate_b      = (const float*)d_in[6];
    const float* pos_bias    = (const float*)d_in[7];
    const float* scale_embed = (const float*)d_in[8];
    const float* if_gain     = (const float*)d_in[9];

    const int    M  = BB * NN;          // 8192
    const size_t nx = (size_t)M * DD;

    char* ws = (char*)d_ws;
    auto carve = [&](size_t bytes) {
        char* p = ws;
        ws += (bytes + 255) & ~(size_t)255;
        return p;
    };
    __hip_bfloat16* x_bf     = (__hip_bfloat16*)carve(nx * 2);
    __hip_bfloat16* qkvw_bf  = (__hip_bfloat16*)carve((size_t)3 * DD * DD * 2);
    __hip_bfloat16* gatew_bf = (__hip_bfloat16*)carve((size_t)DD * DD * 2);
    __hip_bfloat16* outw_bf  = (__hip_bfloat16*)carve((size_t)DD * DD * 2);
    __hip_bfloat16* w2_bf    = (__hip_bfloat16*)carve((size_t)NQSP * DD * 2);
    float*          bias2    = (float*)carve((size_t)NQSP * 4);
    float*          qkv_f    = (float*)carve((size_t)M * 3 * DD * 4);
    float*          qs_f     = (float*)carve((size_t)M * NQSP * 4);
    float*          attn_f   = (float*)carve((size_t)M * DD * 4);
    __hip_bfloat16* y_bf     = (__hip_bfloat16*)carve(nx * 2);

    const int thr = 256;
    cvt_f32_bf16<<<(int)((nx + thr - 1) / thr), thr, 0, stream>>>(x, x_bf, (long)nx);
    cvt_f32_bf16<<<(int)(((size_t)3 * DD * DD + thr - 1) / thr), thr, 0, stream>>>(
        qkv_w, qkvw_bf, (long)3 * DD * DD);
    cvt_f32_bf16<<<(int)(((size_t)DD * DD + thr - 1) / thr), thr, 0, stream>>>(
        gate_w, gatew_bf, (long)DD * DD);
    cvt_f32_bf16<<<(int)(((size_t)DD * DD + thr - 1) / thr), thr, 0, stream>>>(
        out_w, outw_bf, (long)DD * DD);
    build_w2<<<(NQSP * DD + thr - 1) / thr, thr, 0, stream>>>(qkv_w, scale_embed, w2_bf);
    build_bias2<<<(NQSP + thr - 1) / thr, thr, 0, stream>>>(qkv_b, scale_embed, bias2);

    dim3 blk(256);
    // QKV projection: [8192,1024] x [3072,1024]^T
    dim3 gQKV(3 * DD / 128, M / 128);
    gemm_bf16_wmma<0><<<gQKV, blk, LDS_TOTAL_BYTES, stream>>>(
        x_bf, qkvw_bf, qkv_b, qkv_f, nullptr, nullptr, M, 3 * DD, DD);
    // qs = x @ W2^T + bias2 : [8192,1024] x [768,1024]^T (64 zero-pad rows)
    dim3 gQS(NQSP / 128, M / 128);
    gemm_bf16_wmma<0><<<gQS, blk, LDS_TOTAL_BYTES, stream>>>(
        x_bf, w2_bf, bias2, qs_f, nullptr, nullptr, M, NQSP, DD);
    // attention core
    attn_dyadic<<<(BB * HH * NN) / 8, blk, 0, stream>>>(qkv_f, qs_f, pos_bias,
                                                        if_gain, attn_f);
    // gate GEMM fused with sigmoid * attn -> bf16 y
    dim3 gD(DD / 128, M / 128);
    gemm_bf16_wmma<1><<<gD, blk, LDS_TOTAL_BYTES, stream>>>(
        x_bf, gatew_bf, gate_b, nullptr, attn_f, y_bf, M, DD, DD);
    // output projection -> d_out (fp32)
    gemm_bf16_wmma<0><<<gD, blk, LDS_TOTAL_BYTES, stream>>>(
        y_bf, outw_bf, out_b, (float*)d_out, nullptr, nullptr, M, DD, DD);
}
